// NEG_28123445854589
// MI455X (gfx1250) — compile-verified, matrix operand order checked by reading, not run on hardware
//
#include <hip/hip_runtime.h>
#include <hip/hip_bf16.h>
#include <math.h>

#define HDIM 128
#define CDIM 10
#define PDEC 3

typedef __attribute__((ext_vector_type(16))) __bf16 v16bf;
typedef __attribute__((ext_vector_type(8)))  float  v8f;

struct Scalars { int done; float lsum; float cnt; int pad; };

__device__ __forceinline__ unsigned short f32_to_bf16u(float f) {
  unsigned u = __float_as_uint(f);
  u += 0x7FFFu + ((u >> 16) & 1u);          // round-to-nearest-even
  return (unsigned short)(u >> 16);
}
// Order-preserving float -> uint encoding (monotone total order), so that
// unsigned atomicMax == float max. encode(-inf) = 0x007FFFFF.
__device__ __forceinline__ unsigned ford_enc(float f) {
  unsigned u = __float_as_uint(f);
  return (u & 0x80000000u) ? ~u : (u | 0x80000000u);
}
__device__ __forceinline__ float ford_dec(unsigned o) {
  unsigned u = (o & 0x80000000u) ? (o & 0x7FFFFFFFu) : ~o;
  return __uint_as_float(u);
}

// ---------------------------------------------------------------------------
// Generic [rows,128] x [128,128] GEMM using v_wmma_f32_16x16x32_bf16.
// Block = 128 threads = 4 waves; block handles 64 rows; each wave one 16-row
// stripe across all 8 column tiles. A is staged via LDS directly in the
// per-lane WMMA fragment layout (16-bit A 16x32, ISA 7.12.2). B is read from
// a pre-swizzled bf16 buffer: Bsw[((nt*4+kt)*32+lane)*16 + j].
// RELU_COLSUM: instead of storing, relu + atomic column-sum into dst[128].
// ---------------------------------------------------------------------------
template<bool GUARD, bool HAS_AGG, bool RELU_COLSUM>
__global__ __launch_bounds__(128)
void gemm128_kernel(const Scalars* __restrict__ sc,
                    const float* __restrict__ src,
                    const unsigned* __restrict__ aggu,
                    const unsigned short* __restrict__ Bsw,
                    const float* __restrict__ bias,
                    float* __restrict__ dst,
                    int n)
{
  if (GUARD && sc->done) return;
  __shared__ unsigned short aLDS[4 * 4 * 32 * 16];   // 16 KB
  const int row0 = blockIdx.x * 64;

  for (int idx = threadIdx.x; idx < 64 * HDIM; idx += 128) {
    int ml = idx >> 7, k = idx & 127;
    int row = row0 + ml;
    float v = 0.f;
    if (row < n) {
      v = src[(size_t)row * HDIM + k];
      if (HAS_AGG) {
        float a = ford_dec(aggu[(size_t)row * HDIM + k]);
        if (__builtin_isfinite(a)) v += a;           // empty segment (-inf) -> 0
      }
    }
    int wave = ml >> 4, m = ml & 15;
    int kt = k >> 5, kk = k & 31;
    int hi = (kk >> 3) & 1;
    int j  = (kk & 7) | (((kk >> 4) & 1) << 3);
    aLDS[(((wave << 2) | kt) * 32 + (m | (hi << 4))) * 16 + j] = f32_to_bf16u(v);
  }
  __syncthreads();

  const int wave = threadIdx.x >> 5, lane = threadIdx.x & 31;
  const int col = lane & 15, hi = lane >> 4;
  const int wrow0 = row0 + wave * 16;

  v16bf afrag[4];
  for (int kt = 0; kt < 4; ++kt)
    afrag[kt] = *reinterpret_cast<const v16bf*>(&aLDS[(((wave << 2) | kt) * 32 + lane) * 16]);

  for (int nt = 0; nt < 8; ++nt) {
    v8f c = {};
    for (int kt = 0; kt < 4; ++kt) {
      v16bf b = *reinterpret_cast<const v16bf*>(&Bsw[(((nt << 2) | kt) * 32 + lane) * 16]);
      c = __builtin_amdgcn_wmma_f32_16x16x32_bf16(false, afrag[kt], false, b,
                                                  (short)0, c, false, false);
    }
    const float bv = bias[nt * 16 + col];
    if (RELU_COLSUM) {
      float s = 0.f;
      for (int r = 0; r < 8; ++r) {
        int row = wrow0 + r + 8 * hi;
        float u = c[r] + bv;
        u = u > 0.f ? u : 0.f;
        if (row < n) s += u;
      }
      s += __shfl_xor(s, 16);                        // combine the two row-halves
      if (lane < 16) atomicAdd(&dst[nt * 16 + col], s);
    } else {
      for (int r = 0; r < 8; ++r) {
        int row = wrow0 + r + 8 * hi;
        if (row < n) dst[(size_t)row * HDIM + nt * 16 + col] = c[r] + bv;
      }
    }
  }
}

// Decoder: logits[p,n,c] = h @ dec_W[p] + dec_b[p], then log_softmax over C.
// One 16-col WMMA tile per wave (cols 10..15 padded with zero weights and
// -inf bias so the cross-lane softmax reduction is exact).
__global__ __launch_bounds__(128)
void decoder_kernel(const float* __restrict__ h,
                    const unsigned short* __restrict__ decsw, // [P][4][32][16]
                    const float* __restrict__ dec_b,          // [P][10]
                    float* __restrict__ out, int n)
{
  __shared__ unsigned short aLDS[4 * 4 * 32 * 16];
  const int row0 = blockIdx.x * 64;
  const int p = blockIdx.y;

  for (int idx = threadIdx.x; idx < 64 * HDIM; idx += 128) {
    int ml = idx >> 7, k = idx & 127;
    int row = row0 + ml;
    float v = (row < n) ? h[(size_t)row * HDIM + k] : 0.f;
    int wave = ml >> 4, m = ml & 15;
    int kt = k >> 5, kk = k & 31;
    int hi = (kk >> 3) & 1;
    int j  = (kk & 7) | (((kk >> 4) & 1) << 3);
    aLDS[(((wave << 2) | kt) * 32 + (m | (hi << 4))) * 16 + j] = f32_to_bf16u(v);
  }
  __syncthreads();

  const int wave = threadIdx.x >> 5, lane = threadIdx.x & 31;
  const int col = lane & 15, hi = lane >> 4;
  const int wrow0 = row0 + wave * 16;

  v8f c = {};
  for (int kt = 0; kt < 4; ++kt) {
    v16bf a = *reinterpret_cast<const v16bf*>(&aLDS[(((wave << 2) | kt) * 32 + lane) * 16]);
    v16bf b = *reinterpret_cast<const v16bf*>(&decsw[((p * 4 + kt) * 32 + lane) * 16]);
    c = __builtin_amdgcn_wmma_f32_16x16x32_bf16(false, a, false, b, (short)0, c, false, false);
  }
  const float bv = (col < CDIM) ? dec_b[p * CDIM + col] : -INFINITY;
  for (int r = 0; r < 8; ++r) {
    float v = c[r] + bv;
    float m = v;
    for (int off = 1; off < 16; off <<= 1) m = fmaxf(m, __shfl_xor(m, off));
    float e = (col < CDIM) ? __expf(v - m) : 0.f;   // padded lanes contribute 0
    float s = e;
    for (int off = 1; off < 16; off <<= 1) s += __shfl_xor(s, off);
    float lsm = v - m - __logf(s);
    int row = wrow0 + r + 8 * hi;
    if (row < n && col < CDIM)
      out[((size_t)p * n + row) * CDIM + col] = lsm;
  }
}

// -------------------------- small helper kernels ---------------------------
__global__ void init_scalars_kernel(Scalars* sc) {
  sc->done = 0; sc->lsum = 0.f; sc->cnt = 0.f; sc->pad = 0;
}

// Swizzle three 128x128 f32 weights into bf16 WMMA-B fragment layout.
__global__ void swizzle3_kernel(const float* W0, const float* W1, const float* W2,
                                unsigned short* o0, unsigned short* o1, unsigned short* o2)
{
  int idx = blockIdx.x * blockDim.x + threadIdx.x;
  if (idx >= HDIM * HDIM) return;
  const float* W = (blockIdx.y == 0) ? W0 : ((blockIdx.y == 1) ? W1 : W2);
  unsigned short* o = (blockIdx.y == 0) ? o0 : ((blockIdx.y == 1) ? o1 : o2);
  int j = idx & 15, lane = (idx >> 4) & 31, kt = (idx >> 9) & 3, nt = idx >> 11;
  int hi = lane >> 4, col = lane & 15;
  int k = kt * 32 + ((j & 7) | (hi << 3) | (((j >> 3) & 1) << 4));
  o[idx] = f32_to_bf16u(W[k * HDIM + nt * 16 + col]);
}

// Swizzle dec_W [P,128,10] into [P][4][32][16] bf16 fragments (cols >=10 zero).
__global__ void swizzle_dec_kernel(const float* decW, unsigned short* o) {
  int idx = blockIdx.x * blockDim.x + threadIdx.x;
  if (idx >= PDEC * 4 * 32 * 16) return;
  int j = idx & 15, lane = (idx >> 4) & 31, kt = (idx >> 9) & 3, p = idx >> 11;
  int hi = lane >> 4, col = lane & 15;
  int k = kt * 32 + ((j & 7) | (hi << 3) | (((j >> 3) & 1) << 4));
  float v = (col < CDIM) ? decW[((size_t)p * HDIM + k) * CDIM + col] : 0.f;
  o[idx] = f32_to_bf16u(v);
}

__global__ void fill_agg_kernel(const Scalars* sc, unsigned* aggu, size_t total) {
  if (sc->done) return;
  size_t i = (size_t)blockIdx.x * blockDim.x + threadIdx.x;
  if (i < total) aggu[i] = 0x007FFFFFu;             // ford_enc(-inf)
}

// segment_max: 32 lanes per edge, float4 per lane -> native u32 max atomics.
__global__ void scatter_max_kernel(const Scalars* __restrict__ sc,
                                   const int* __restrict__ ei,
                                   const float* __restrict__ h,
                                   unsigned* __restrict__ aggu, int ecount)
{
  if (sc->done) return;
  long long g = (long long)blockIdx.x * blockDim.x + threadIdx.x;
  if (g >= (long long)ecount * 32) return;
  int e  = (int)(g >> 5);
  int k4 = (int)(g & 31) << 2;
  int s = ei[e], d = ei[ecount + e];
  const float4 hv = *reinterpret_cast<const float4*>(h + (size_t)s * HDIM + k4);
  unsigned* b = aggu + (size_t)d * HDIM + k4;
  atomicMax(b + 0, ford_enc(hv.x));
  atomicMax(b + 1, ford_enc(hv.y));
  atomicMax(b + 2, ford_enc(hv.z));
  atomicMax(b + 3, ford_enc(hv.w));
}

__global__ void zero_colsum_kernel(float* colsum) { colsum[threadIdx.x] = 0.f; }

// mean over nodes -> t1_W2 -> t2 -> sigmoid -> BCE; latch done (p > 0.9).
__global__ void finish_step_kernel(Scalars* sc, const float* colsum,
                                   const float* t1W2, const float* t1b2,
                                   const float* t2W, const float* t2b,
                                   const int* diam, int step, int n)
{
  __shared__ float mean[HDIM];
  __shared__ float red[HDIM];
  if (sc->done) return;
  int tid = threadIdx.x;
  mean[tid] = colsum[tid] / (float)n;
  __syncthreads();
  float acc = t1b2[tid];
  for (int k = 0; k < HDIM; ++k) acc += mean[k] * t1W2[k * HDIM + tid];
  red[tid] = acc * t2W[tid];
  __syncthreads();
  for (int s = 64; s > 0; s >>= 1) {
    if (tid < s) red[tid] += red[tid + s];
    __syncthreads();
  }
  if (tid == 0) {
    float z = red[0] + t2b[0];
    float p = 1.f / (1.f + __expf(-z));
    float y = ((float)(step + 1) >= (float)(*diam)) ? 1.f : 0.f;
    float loss = -(y * __logf(p) + (1.f - y) * __logf(1.f - p));
    sc->lsum += loss;
    sc->cnt  += 1.f;
    if (p > 0.9f) sc->done = 1;
  }
}

__global__ void write_aux_kernel(const Scalars* sc, float* out, size_t pos) {
  out[pos] = sc->lsum / sc->cnt;
}

// ---------------------------------------------------------------------------
extern "C" void kernel_launch(void* const* d_in, const int* in_sizes, int n_in,
                              void* d_out, int out_size, void* d_ws, size_t ws_size,
                              hipStream_t stream)
{
  const float* x     = (const float*)d_in[0];
  const int*   eidx  = (const int*)d_in[1];
  const int*   diam  = (const int*)d_in[2];
  const float* enc_W = (const float*)d_in[3];
  const float* enc_b = (const float*)d_in[4];
  const float* gin_W = (const float*)d_in[5];
  const float* gin_b = (const float*)d_in[6];
  const float* t1_W1 = (const float*)d_in[7];
  const float* t1_b1 = (const float*)d_in[8];
  const float* t1_W2 = (const float*)d_in[9];
  const float* t1_b2 = (const float*)d_in[10];
  const float* t2_W  = (const float*)d_in[11];
  const float* t2_b  = (const float*)d_in[12];
  const float* dec_W = (const float*)d_in[13];
  const float* dec_b = (const float*)d_in[14];
  (void)n_in; (void)ws_size;

  const int n      = in_sizes[0] / HDIM;
  const int ecount = in_sizes[1] / 2;

  char* ws = (char*)d_ws;
  size_t off = 0;
  auto alloc = [&](size_t bytes) -> void* {
    void* p = ws + off; off += (bytes + 255) & ~(size_t)255; return p;
  };
  float*          h      = (float*)alloc((size_t)n * HDIM * sizeof(float));
  unsigned*       aggu   = (unsigned*)alloc((size_t)n * HDIM * sizeof(unsigned));
  unsigned short* enc_sw = (unsigned short*)alloc((size_t)HDIM * HDIM * 2);
  unsigned short* gin_sw = (unsigned short*)alloc((size_t)HDIM * HDIM * 2);
  unsigned short* t1_sw  = (unsigned short*)alloc((size_t)HDIM * HDIM * 2);
  unsigned short* dec_sw = (unsigned short*)alloc((size_t)PDEC * 4 * 32 * 16 * 2);
  float*          colsum = (float*)alloc(HDIM * sizeof(float));
  Scalars*        sc     = (Scalars*)alloc(sizeof(Scalars));

  init_scalars_kernel<<<1, 1, 0, stream>>>(sc);
  swizzle3_kernel<<<dim3((HDIM * HDIM + 255) / 256, 3), 256, 0, stream>>>(
      enc_W, gin_W, t1_W1, enc_sw, gin_sw, t1_sw);
  swizzle_dec_kernel<<<(PDEC * 4 * 32 * 16 + 255) / 256, 256, 0, stream>>>(dec_W, dec_sw);

  const int gemmBlocks = (n + 63) / 64;

  // encoder: h = x @ enc_W + enc_b   (unguarded, no aggregation)
  gemm128_kernel<false, false, false><<<gemmBlocks, 128, 0, stream>>>(
      sc, x, nullptr, enc_sw, enc_b, h, n);

  const size_t aggTotal = (size_t)n * HDIM;
  const int aggBlocks  = (int)((aggTotal + 255) / 256);
  const int scatBlocks = (int)(((long long)ecount * 32 + 255) / 256);

  // T = min(2*diameter, N); reference makes this static (diameter = 6 -> 12).
  int T = 12 < n ? 12 : n;
  for (int step = 0; step < T; ++step) {
    fill_agg_kernel<<<aggBlocks, 256, 0, stream>>>(sc, aggu, aggTotal);
    scatter_max_kernel<<<scatBlocks, 256, 0, stream>>>(sc, eidx, h, aggu, ecount);
    // h = (h + agg) @ gin_W + gin_b   (in-place: block stages its rows first)
    gemm128_kernel<true, true, false><<<gemmBlocks, 128, 0, stream>>>(
        sc, h, aggu, gin_sw, gin_b, h, n);
    zero_colsum_kernel<<<1, HDIM, 0, stream>>>(colsum);
    // colsum = sum_n relu(h @ t1_W1 + t1_b1)
    gemm128_kernel<true, false, true><<<gemmBlocks, 128, 0, stream>>>(
        sc, h, nullptr, t1_sw, t1_b1, colsum, n);
    finish_step_kernel<<<1, HDIM, 0, stream>>>(
        sc, colsum, t1_W2, t1_b2, t2_W, t2_b, diam, step, n);
  }

  decoder_kernel<<<dim3(gemmBlocks, PDEC), 128, 0, stream>>>(
      h, dec_sw, dec_b, (float*)d_out, n);
  write_aux_kernel<<<1, 1, 0, stream>>>(sc, (float*)d_out, (size_t)out_size - 1);
}
